// MLP_head_79336635892046
// MI455X (gfx1250) — matrix operand order repaired and checked by reading, not compiled, not run on hardware
//
#include <hip/hip_runtime.h>
#include <hip/hip_bf16.h>

// ---------------------------------------------------------------------------
// MLP head: per-class [128x1024]@[1024x1024] GEMM + bias + exact GELU + dot(W2)
// One workgroup per (batch, class). x tile lives fully in LDS (CDNA5 320KB WGP).
// bf16 WMMA (v_wmma_f32_16x16x32_bf16), fp32->bf16 converted on the fly.
// W1 double-buffered; prefetch issued AFTER the barrier so global-load latency
// hides behind the WMMA block instead of stalling at the barrier.
// ---------------------------------------------------------------------------

typedef __attribute__((ext_vector_type(16))) __bf16        v16bf;
typedef __attribute__((ext_vector_type(8)))  float         v8f;
typedef __attribute__((ext_vector_type(4)))  unsigned int  u32x4;
typedef __attribute__((ext_vector_type(2)))  unsigned int  u32x2;

#define NUM_CLASSES 64
#define CDF_DIM     128
#define EMBED       1024
#define BATCH       16

#define XPITCH      1032      // bf16 elems per x row in LDS (1024 + 8 pad)
#define KT          64        // K chunk staged per step
#define WPITCH      72        // bf16 elems per W1-tile row (64 + 8 pad)
#define NT          128       // N tile width
#define KSTEPS      (EMBED / KT)

#define XBYTES      (CDF_DIM * XPITCH * 2)          // 264192
#define WBYTES      (2 * NT * WPITCH * 2)           //  36864
#define SMEM_BYTES  (XBYTES + WBYTES + EMBED*4 + EMBED*4 + CDF_DIM*4)  // 309760

__device__ __forceinline__ unsigned f2bf(float f) {
    unsigned u = __float_as_uint(f);
    unsigned r = u + 0x7FFFu + ((u >> 16) & 1u);    // round-to-nearest-even
    return r >> 16;
}

__device__ __forceinline__ unsigned pk_bf2(float a, float b) {
    return f2bf(a) | (f2bf(b) << 16);
}

// Build a 16-element bf16 fragment from two 16-byte LDS chunks (ds_load_b128 x2)
__device__ __forceinline__ v16bf ld_frag(const unsigned short* p, int e1) {
    union { u32x4 u[2]; v16bf v; } t;
    t.u[0] = *(const u32x4*)(p);
    t.u[1] = *(const u32x4*)(p + e1);
    return t.v;
}

__global__ __launch_bounds__(256)
void mlp_head_kernel(const float* __restrict__ x,  const float* __restrict__ W1,
                     const float* __restrict__ b1, const float* __restrict__ W2,
                     const float* __restrict__ b2, float* __restrict__ out)
{
    extern __shared__ unsigned char smem_raw[];
    unsigned short* xs   = (unsigned short*)smem_raw;            // [128][XPITCH] bf16
    unsigned short* w1t0 = xs + CDF_DIM * XPITCH;                // [NT][WPITCH] bf16
    unsigned short* w1t1 = w1t0 + NT * WPITCH;
    float* b1s  = (float*)(w1t1 + NT * WPITCH);                  // [1024]
    float* w2s  = b1s + EMBED;                                   // [1024]
    float* oacc = w2s + EMBED;                                   // [128]

    const int c    = blockIdx.x;
    const int b    = blockIdx.y;
    const int tid  = threadIdx.x;
    const int lane = tid & 31;
    const int wave = tid >> 5;
    const int wm   = wave & 3;    // 4 M-blocks of 32 rows
    const int wn   = wave >> 2;   // 2 N-blocks of 64 cols (within 128-wide tile)

    const float* xg  = x  + (size_t)(b * (NUM_CLASSES * CDF_DIM) + c * CDF_DIM) * EMBED;
    const float* w1g = W1 + (size_t)c * EMBED * EMBED;

    // W1 staging map: each thread owns one column nn and 32 *contiguous* k rows,
    // so LDS writes pack 8 bf16 into one ds_store_b128.
    const int whi = tid >> 7;          // 0/1 -> k rows [0..31] / [32..63]
    const int wnn = tid & 127;         // column within 128-wide N tile

    // ---- stage x tile (fp32 -> bf16, float4 loads / b64 packed stores) ----
    for (int i = tid; i < CDF_DIM * (EMBED / 4); i += 256) {
        int row = i >> 8;              // 256 quads per row
        int q   = i & 255;
        float4 v = *(const float4*)(xg + row * EMBED + q * 4);
        u32x2 pk;
        pk.x = pk_bf2(v.x, v.y);
        pk.y = pk_bf2(v.z, v.w);
        *(u32x2*)(xs + row * XPITCH + q * 4) = pk;
    }
    for (int i = tid; i < EMBED; i += 256) {
        b1s[i] = b1[c * EMBED + i];
        w2s[i] = W2[c * EMBED + i];
    }
    if (tid < CDF_DIM) oacc[tid] = 0.0f;
    __syncthreads();

    float rowpart[2][8];
    #pragma unroll
    for (int mi = 0; mi < 2; ++mi)
        #pragma unroll
        for (int r = 0; r < 8; ++r) rowpart[mi][r] = 0.0f;

    const int arow0 = wm * 32 + (lane & 15);        // A frag row (per mi: +16)
    const int akoff = (lane >> 4) * 16;             // A: lane-halves contiguous 16
    const int bkoff = (lane >> 4) * 16;             // B: lane-halves contiguous 16

    for (int n0 = 0; n0 < EMBED; n0 += NT) {
        v8f acc[2][4];
        #pragma unroll
        for (int mi = 0; mi < 2; ++mi)
            #pragma unroll
            for (int ni = 0; ni < 4; ++ni)
                #pragma unroll
                for (int r = 0; r < 8; ++r) acc[mi][ni][r] = 0.0f;

        // prologue: load first 64x128 W1 chunk (32 fp32 per thread)
        float wreg[32];
        #pragma unroll
        for (int j = 0; j < 32; ++j)
            wreg[j] = w1g[(size_t)(whi * 32 + j) * EMBED + n0 + wnn];

        for (int kt = 0; kt < KSTEPS; ++kt) {
            unsigned short* wb = (kt & 1) ? w1t1 : w1t0;
            // convert + pack 8 bf16 -> one b128 LDS store (4 per thread)
            #pragma unroll
            for (int g = 0; g < 4; ++g) {
                u32x4 pk;
                #pragma unroll
                for (int h = 0; h < 4; ++h)
                    pk[h] = pk_bf2(wreg[g * 8 + h * 2], wreg[g * 8 + h * 2 + 1]);
                *(u32x4*)(wb + wnn * WPITCH + whi * 32 + g * 8) = pk;
            }
            __syncthreads();   // needs only dscnt==0 before signal (double-buffered)

            // issue next chunk's global loads AFTER the barrier: their latency
            // is hidden by the WMMA block below, and the wait lands at the next
            // iteration's pack (post-compute), not at the barrier.
            if (kt + 1 < KSTEPS) {
                #pragma unroll
                for (int j = 0; j < 32; ++j)
                    wreg[j] = w1g[(size_t)((kt + 1) * KT + whi * 32 + j) * EMBED + n0 + wnn];
            }

            #pragma unroll
            for (int ks = 0; ks < KT; ks += 32) {
                const int kbase = kt * KT + ks;
                v16bf afrag[2];
                #pragma unroll
                for (int mi = 0; mi < 2; ++mi)
                    afrag[mi] = ld_frag(xs + (arow0 + mi * 16) * XPITCH + kbase + akoff, 8);
                #pragma unroll
                for (int ni = 0; ni < 4; ++ni) {
                    v16bf bfrag = ld_frag(wb + (wn * 64 + ni * 16 + (lane & 15)) * WPITCH
                                             + ks + bkoff, 8);
                    #pragma unroll
                    for (int mi = 0; mi < 2; ++mi)
                        acc[mi][ni] = __builtin_amdgcn_wmma_f32_16x16x32_bf16(
                            false, afrag[mi], false, bfrag,
                            (short)0, acc[mi][ni], false, false);
                }
            }
        }

        // fused epilogue for this N-tile: +b1, exact GELU, * W2, row-accumulate
        #pragma unroll
        for (int ni = 0; ni < 4; ++ni) {
            int n = n0 + wn * 64 + ni * 16 + (lane & 15);
            float bb = b1s[n], ww = w2s[n];
            #pragma unroll
            for (int mi = 0; mi < 2; ++mi)
                #pragma unroll
                for (int r = 0; r < 8; ++r) {
                    float h = acc[mi][ni][r] + bb;
                    float g = 0.5f * h * (1.0f + erff(h * 0.70710678118654752f));
                    rowpart[mi][r] += g * ww;
                }
        }
    }

    // reduce per-row partials across lanes/waves via LDS atomics
    #pragma unroll
    for (int mi = 0; mi < 2; ++mi)
        #pragma unroll
        for (int r = 0; r < 8; ++r) {
            int m = wm * 32 + mi * 16 + ((lane >> 4) << 3) + r;
            atomicAdd(&oacc[m], rowpart[mi][r]);
        }
    __syncthreads();

    if (tid < CDF_DIM)
        out[(size_t)b * (NUM_CLASSES * CDF_DIM) + c * CDF_DIM + tid] = oacc[tid] + b2[c];
}

extern "C" void kernel_launch(void* const* d_in, const int* in_sizes, int n_in,
                              void* d_out, int out_size, void* d_ws, size_t ws_size,
                              hipStream_t stream) {
    const float* x  = (const float*)d_in[0];
    const float* W1 = (const float*)d_in[1];
    const float* b1 = (const float*)d_in[2];
    const float* W2 = (const float*)d_in[3];
    const float* b2 = (const float*)d_in[4];
    float* out = (float*)d_out;

    (void)in_sizes; (void)n_in; (void)out_size; (void)d_ws; (void)ws_size;

    hipFuncSetAttribute((const void*)mlp_head_kernel,
                        hipFuncAttributeMaxDynamicSharedMemorySize, SMEM_BYTES);

    mlp_head_kernel<<<dim3(NUM_CLASSES, BATCH), dim3(256), SMEM_BYTES, stream>>>(
        x, W1, b1, W2, b2, out);
}
